// SparseMoEBlock_24799141167303
// MI455X (gfx1250) — compile-verified, matrix operand order checked by reading, not run on hardware
//
#include <hip/hip_runtime.h>
#include <cstdint>

// ---------------- problem constants ----------------
#define NE 8        // experts
#define KK 2        // top-k
#define DD 1024     // embed dim
#define FF 4096     // expert intermediate
#define FSH 2048    // shared intermediate
#define NT 2048     // tokens (B*S)

// ---------------- tile constants ----------------
#define BM 64       // rows per block
#define BN 128      // cols per block
#define BK 32       // k per step
#define LDA 40      // LDS row stride in bf16 elems (80B: 16B-aligned frags)

// ---------------- types ----------------
typedef __attribute__((ext_vector_type(16))) __bf16 v16bf;
typedef __attribute__((ext_vector_type(8)))  __bf16 v8bf;
typedef __attribute__((ext_vector_type(4)))  __bf16 v4bf;
typedef __attribute__((ext_vector_type(8)))  float  v8f;
typedef __attribute__((ext_vector_type(4)))  float  fx4;

__device__ __forceinline__ v8f wmma_bf16(v16bf a, v16bf b, v8f c) {
  return __builtin_amdgcn_wmma_f32_16x16x32_bf16(false, a, false, b,
                                                 (short)0, c, false, false);
}

// 16-bit A/B 16x32 fragment per ISA layout: lane<16 takes K[0..7],[16..23],
// lane>=16 takes K[8..15],[24..31]; p = row base + (lane>>4)*8 elems.
__device__ __forceinline__ v16bf load_frag(const __bf16* p) {
  v8bf lo = *(const v8bf*)p;
  v8bf hi = *(const v8bf*)(p + 16);
  return __builtin_shufflevector(lo, hi, 0, 1, 2, 3, 4, 5, 6, 7,
                                 8, 9, 10, 11, 12, 13, 14, 15);
}

__device__ __forceinline__ float silu_fast(float g) {
  return g * __builtin_amdgcn_rcpf(1.0f + __expf(-g));
}

// ---------------- kernel: zero counters ----------------
__global__ void moe_zero_cnt(int* cnt) {
  if (threadIdx.x < NE) cnt[threadIdx.x] = 0;
}

// ---------------- kernel: router (one wave per token) ----------------
__global__ __launch_bounds__(256) void moe_router(const float* __restrict__ x,
                                                  const float* __restrict__ gw,
                                                  int* cnt, int* lists,
                                                  float* wslot) {
  int wave = threadIdx.x >> 5;
  int lane = threadIdx.x & 31;
  int tok = blockIdx.x * 8 + wave;
  if (tok >= NT) return;

  float acc[NE];
#pragma unroll
  for (int e = 0; e < NE; ++e) acc[e] = 0.0f;

  const float* xr = x + (size_t)tok * DD;
  for (int d = lane; d < DD; d += 32) {
    float xv = xr[d];
#pragma unroll
    for (int e = 0; e < NE; ++e) acc[e] += xv * gw[e * DD + d];
  }
#pragma unroll
  for (int e = 0; e < NE; ++e) {
#pragma unroll
    for (int off = 16; off > 0; off >>= 1)
      acc[e] += __shfl_xor(acc[e], off, 32);
  }
  if (lane != 0) return;

  float m = acc[0];
#pragma unroll
  for (int e = 1; e < NE; ++e) m = fmaxf(m, acc[e]);
  float p[NE], se = 0.0f;
#pragma unroll
  for (int e = 0; e < NE; ++e) { p[e] = __expf(acc[e] - m); se += p[e]; }
  float inv = __builtin_amdgcn_rcpf(se);
#pragma unroll
  for (int e = 0; e < NE; ++e) p[e] *= inv;

  int i0 = 0;
#pragma unroll
  for (int e = 1; e < NE; ++e) if (p[e] > p[i0]) i0 = e;
  int i1 = (i0 == 0) ? 1 : 0;
#pragma unroll
  for (int e = 0; e < NE; ++e)
    if (e != i0 && p[e] > p[i1]) i1 = e;

  int s0 = tok * KK + 0, s1 = tok * KK + 1;
  int pos0 = atomicAdd(&cnt[i0], 1);
  lists[i0 * NT + pos0] = s0;
  wslot[s0] = p[i0];
  int pos1 = atomicAdd(&cnt[i1], 1);
  lists[i1 * NT + pos1] = s1;
  wslot[s1] = p[i1];
}

// ---------------- kernel: x -> bf16 ----------------
__global__ __launch_bounds__(256) void moe_cvt_x(const float* __restrict__ x,
                                                 __bf16* __restrict__ xb) {
  int i = (blockIdx.x * 256 + threadIdx.x) * 4;
  fx4 v = *(const fx4*)(x + i);
  v4bf o = {(__bf16)v.x, (__bf16)v.y, (__bf16)v.z, (__bf16)v.w};
  *(v4bf*)(xb + i) = o;
}

// =====================================================================
// Fused gate+up SwiGLU GEMM: act = silu(A@Wg) * (A@Wu)
// A rows: gathered tokens (GATHER) or direct tokens. K = DD. B ld = LDW.
// Block: 256 thr (8 waves, 2M x 4N), tile 64x128, double-buffered LDS.
// grid: (LDW/BN, NT/BM, experts)
// =====================================================================
template <int LDW, bool GATHER>
__global__ __launch_bounds__(256) void moe_gateup_k(
    const __bf16* __restrict__ xb, const int* __restrict__ cnt,
    const int* __restrict__ lists, const float* __restrict__ WgB,
    const float* __restrict__ WuB, __bf16* __restrict__ actOut) {
  __shared__ __align__(16) __bf16 As[2][BM * LDA];
  __shared__ __align__(16) __bf16 Bg[2][BN * LDA];
  __shared__ __align__(16) __bf16 Bu[2][BN * LDA];
  __shared__ int slotIds[BM];

  const int e = blockIdx.z;
  const int mBase = blockIdx.y * BM;
  int c = NT;
  if (GATHER) {
    c = cnt[e];
    if (mBase >= c) return;
  }
  const int n0 = blockIdx.x * BN;
  const float* Wg = WgB + (GATHER ? (size_t)e * DD * LDW : 0);
  const float* Wu = WuB + (GATHER ? (size_t)e * DD * LDW : 0);

  const int tid = threadIdx.x;
  const int lane = tid & 31, wave = tid >> 5;
  const int mw = wave & 1, nw = wave >> 1;
  const int half = lane >> 4, r15 = lane & 15;

  if (GATHER) {
    if (tid < BM) {
      int m = mBase + tid;
      slotIds[tid] = (m < c) ? lists[e * NT + m] : -1;
    }
    __syncthreads();
  }

  // staging indices
  const int ar = tid >> 2, ak = (tid & 3) * 8;          // A: 1 b128 / thread
  const int bk = (tid & 7) * 4, bn = (tid >> 3) * 4;    // B: 4 fx4 / matrix

  v8bf aReg;
  fx4 rg[4], ru[4];

  auto loadStep = [&](int k0) {
    if (GATHER) {
      int slot = slotIds[ar];
      if (slot >= 0)
        aReg = *(const v8bf*)(xb + (size_t)(slot >> 1) * DD + k0 + ak);
      else
        aReg = (v8bf){};
    } else {
      aReg = *(const v8bf*)(xb + (size_t)(mBase + ar) * DD + k0 + ak);
    }
#pragma unroll
    for (int t = 0; t < 4; ++t) {
      size_t off = (size_t)(k0 + bk + t) * LDW + n0 + bn;
      rg[t] = *(const fx4*)(Wg + off);
      ru[t] = *(const fx4*)(Wu + off);
    }
  };
  auto storeStep = [&](int buf) {
    *(v8bf*)(&As[buf][ar * LDA + ak]) = aReg;
#pragma unroll
    for (int t2 = 0; t2 < 4; ++t2) {
      v4bf vg = {(__bf16)rg[0][t2], (__bf16)rg[1][t2], (__bf16)rg[2][t2],
                 (__bf16)rg[3][t2]};
      v4bf vu = {(__bf16)ru[0][t2], (__bf16)ru[1][t2], (__bf16)ru[2][t2],
                 (__bf16)ru[3][t2]};
      *(v4bf*)(&Bg[buf][(bn + t2) * LDA + bk]) = vg;
      *(v4bf*)(&Bu[buf][(bn + t2) * LDA + bk]) = vu;
    }
  };

  v8f accG[2][2] = {{{}, {}}, {{}, {}}};
  v8f accU[2][2] = {{{}, {}}, {{}, {}}};

  loadStep(0);
  storeStep(0);
  int cur = 0;
  for (int k0 = BK;; k0 += BK) {
    __syncthreads();
    const bool more = (k0 < DD);
    if (more) loadStep(k0);

    v16bf af[2];
#pragma unroll
    for (int ms = 0; ms < 2; ++ms)
      af[ms] = load_frag(&As[cur][(mw * 32 + ms * 16 + r15) * LDA + half * 8]);
#pragma unroll
    for (int ns = 0; ns < 2; ++ns) {
      v16bf bg = load_frag(&Bg[cur][(nw * 32 + ns * 16 + r15) * LDA + half * 8]);
      v16bf bu = load_frag(&Bu[cur][(nw * 32 + ns * 16 + r15) * LDA + half * 8]);
#pragma unroll
      for (int ms = 0; ms < 2; ++ms) {
        accG[ms][ns] = wmma_bf16(af[ms], bg, accG[ms][ns]);
        accU[ms][ns] = wmma_bf16(af[ms], bu, accU[ms][ns]);
      }
    }
    if (!more) break;
    storeStep(cur ^ 1);
    cur ^= 1;
  }

  // epilogue: silu(g)*u -> bf16 act
#pragma unroll
  for (int ms = 0; ms < 2; ++ms) {
#pragma unroll
    for (int ns = 0; ns < 2; ++ns) {
      int col = n0 + nw * 32 + ns * 16 + r15;
#pragma unroll
      for (int r = 0; r < 8; ++r) {
        int rowInTile = mw * 32 + ms * 16 + half * 8 + r;
        float v = silu_fast(accG[ms][ns][r]) * accU[ms][ns][r];
        if (GATHER) {
          int slot = slotIds[rowInTile];
          if (slot >= 0) actOut[(size_t)slot * LDW + col] = (__bf16)v;
        } else {
          actOut[(size_t)(mBase + rowInTile) * LDW + col] = (__bf16)v;
        }
      }
    }
  }
}

// =====================================================================
// Down-projection GEMM: out = A(bf16, stride KD) @ Wd[KD][DD]
// GATHER: rows = act slots, result weighted into downout[slot][DD]
// else:   rows = tokens, result written to out[m][DD]  (overwrites d_out)
// grid: (DD/BN, NT/BM, experts)
// =====================================================================
template <int KD, bool GATHER>
__global__ __launch_bounds__(256) void moe_down_k(
    const __bf16* __restrict__ aSrc, const int* __restrict__ cnt,
    const int* __restrict__ lists, const float* __restrict__ wslot,
    const float* __restrict__ WdB, float* __restrict__ outP) {
  __shared__ __align__(16) __bf16 As[2][BM * LDA];
  __shared__ __align__(16) __bf16 Bs[2][BN * LDA];
  __shared__ int slotIds[BM];

  const int e = blockIdx.z;
  const int mBase = blockIdx.y * BM;
  int c = NT;
  if (GATHER) {
    c = cnt[e];
    if (mBase >= c) return;
  }
  const int n0 = blockIdx.x * BN;
  const float* Wd = WdB + (GATHER ? (size_t)e * KD * DD : 0);

  const int tid = threadIdx.x;
  const int lane = tid & 31, wave = tid >> 5;
  const int mw = wave & 1, nw = wave >> 1;
  const int half = lane >> 4, r15 = lane & 15;

  if (GATHER) {
    if (tid < BM) {
      int m = mBase + tid;
      slotIds[tid] = (m < c) ? lists[e * NT + m] : -1;
    }
    __syncthreads();
  }

  const int ar = tid >> 2, ak = (tid & 3) * 8;
  const int bk = (tid & 7) * 4, bn = (tid >> 3) * 4;

  v8bf aReg;
  fx4 rb[4];

  auto loadStep = [&](int k0) {
    if (GATHER) {
      int slot = slotIds[ar];
      if (slot >= 0)
        aReg = *(const v8bf*)(aSrc + (size_t)slot * KD + k0 + ak);
      else
        aReg = (v8bf){};
    } else {
      aReg = *(const v8bf*)(aSrc + (size_t)(mBase + ar) * KD + k0 + ak);
    }
#pragma unroll
    for (int t = 0; t < 4; ++t)
      rb[t] = *(const fx4*)(Wd + (size_t)(k0 + bk + t) * DD + n0 + bn);
  };
  auto storeStep = [&](int buf) {
    *(v8bf*)(&As[buf][ar * LDA + ak]) = aReg;
#pragma unroll
    for (int t2 = 0; t2 < 4; ++t2) {
      v4bf vb = {(__bf16)rb[0][t2], (__bf16)rb[1][t2], (__bf16)rb[2][t2],
                 (__bf16)rb[3][t2]};
      *(v4bf*)(&Bs[buf][(bn + t2) * LDA + bk]) = vb;
    }
  };

  v8f acc[2][2] = {{{}, {}}, {{}, {}}};

  loadStep(0);
  storeStep(0);
  int cur = 0;
  for (int k0 = BK;; k0 += BK) {
    __syncthreads();
    const bool more = (k0 < KD);
    if (more) loadStep(k0);

    v16bf af[2];
#pragma unroll
    for (int ms = 0; ms < 2; ++ms)
      af[ms] = load_frag(&As[cur][(mw * 32 + ms * 16 + r15) * LDA + half * 8]);
#pragma unroll
    for (int ns = 0; ns < 2; ++ns) {
      v16bf b = load_frag(&Bs[cur][(nw * 32 + ns * 16 + r15) * LDA + half * 8]);
#pragma unroll
      for (int ms = 0; ms < 2; ++ms)
        acc[ms][ns] = wmma_bf16(af[ms], b, acc[ms][ns]);
    }
    if (!more) break;
    storeStep(cur ^ 1);
    cur ^= 1;
  }

#pragma unroll
  for (int ms = 0; ms < 2; ++ms) {
#pragma unroll
    for (int ns = 0; ns < 2; ++ns) {
      int col = n0 + nw * 32 + ns * 16 + r15;
#pragma unroll
      for (int r = 0; r < 8; ++r) {
        int rowInTile = mw * 32 + ms * 16 + half * 8 + r;
        if (GATHER) {
          int slot = slotIds[rowInTile];
          if (slot >= 0)
            outP[(size_t)slot * DD + col] = acc[ms][ns][r] * wslot[slot];
        } else {
          outP[(size_t)(mBase + rowInTile) * DD + col] = acc[ms][ns][r];
        }
      }
    }
  }
}

// ---------------- final combine: out += slot0 + slot1 ----------------------
__global__ __launch_bounds__(256) void moe_combine(
    float* __restrict__ out, const float* __restrict__ downout) {
  int i = (blockIdx.x * 256 + threadIdx.x) * 4;  // i = n*DD + d
  int n = i >> 10;
  int d = i & (DD - 1);
  fx4 a = *(const fx4*)(downout + (size_t)(2 * n) * DD + d);
  fx4 b = *(const fx4*)(downout + (size_t)(2 * n + 1) * DD + d);
  fx4 o = *(const fx4*)(out + i);
  *(fx4*)(out + i) = o + a + b;
}

// ---------------- launcher ----------------
extern "C" void kernel_launch(void* const* d_in, const int* in_sizes, int n_in,
                              void* d_out, int out_size, void* d_ws,
                              size_t ws_size, hipStream_t stream) {
  const float* x      = (const float*)d_in[0];
  const float* gate_w = (const float*)d_in[1];
  const float* gate_k = (const float*)d_in[2];
  const float* up_k   = (const float*)d_in[3];
  const float* down_k = (const float*)d_in[4];
  const float* sh_g   = (const float*)d_in[5];
  const float* sh_u   = (const float*)d_in[6];
  const float* sh_d   = (const float*)d_in[7];
  float* out = (float*)d_out;

  uint8_t* ws = (uint8_t*)d_ws;
  size_t off = 0;
  int* cnt = (int*)(ws + off);              off += 256;
  int* lists = (int*)(ws + off);            off += (size_t)NE * NT * 4;
  float* wslot = (float*)(ws + off);        off += (size_t)NT * KK * 4;
  __bf16* xb = (__bf16*)(ws + off);         off += (size_t)NT * DD * 2;
  __bf16* act = (__bf16*)(ws + off);        off += (size_t)NT * KK * FF * 2;
  __bf16* shact = (__bf16*)(ws + off);      off += (size_t)NT * FSH * 2;
  float* downout = (float*)(ws + off);      off += (size_t)NT * KK * DD * 4;

  moe_zero_cnt<<<1, 32, 0, stream>>>(cnt);
  moe_router<<<NT / 8, 256, 0, stream>>>(x, gate_w, cnt, lists, wslot);
  moe_cvt_x<<<(NT * DD) / 1024, 256, 0, stream>>>(x, xb);

  dim3 blk(256);
  // shared expert: stage1 -> shact, stage2 writes d_out (no zero pass needed)
  moe_gateup_k<FSH, false><<<dim3(FSH / BN, NT / BM, 1), blk, 0, stream>>>(
      xb, nullptr, nullptr, sh_g, sh_u, shact);
  moe_down_k<FSH, false><<<dim3(DD / BN, NT / BM, 1), blk, 0, stream>>>(
      shact, nullptr, nullptr, nullptr, sh_d, out);

  // routed experts (gathered, count-guarded over-launch)
  moe_gateup_k<FF, true><<<dim3(FF / BN, NT / BM, NE), blk, 0, stream>>>(
      xb, cnt, lists, gate_k, up_k, act);
  moe_down_k<FF, true><<<dim3(DD / BN, NT / BM, NE), blk, 0, stream>>>(
      act, cnt, lists, wslot, down_k, downout);

  moe_combine<<<(NT * DD) / 1024, 256, 0, stream>>>(out, downout);
}
